// TransformerEncoder_54975581389213
// MI455X (gfx1250) — matrix operand order repaired and checked by reading, not compiled, not run on hardware
//
#include <hip/hip_runtime.h>
#include <hip/hip_bf16.h>
#include <stdint.h>

// Problem constants (reference: B=8, S=1024, D=1024, H=16, DH=64, L=4, FF=4096)
#define BB   8
#define SS   1024
#define DD   1024
#define HH   16
#define DHD  64
#define FFD  4096
#define LSEL 3          // only the last layer's output is returned (reference bug)

typedef __attribute__((ext_vector_type(16))) __bf16 v16bf;
typedef __attribute__((ext_vector_type(8)))  float  v8f;

union Frag16 { v16bf v; unsigned short u[16]; uint4 q[2]; };

__device__ __forceinline__ unsigned short f2bf(float f) {
  unsigned int u = __float_as_uint(f);
  unsigned int r = u + 0x7FFFu + ((u >> 16) & 1u);   // round-to-nearest-even
  return (unsigned short)(r >> 16);
}

__device__ __forceinline__ v8f zero8() {
  v8f z = {0.f,0.f,0.f,0.f,0.f,0.f,0.f,0.f};
  return z;
}

// A fragment: 16x32 bf16 (MxK).  ISA layout: lanes 0-15 -> M=lane, K={e}(e<8), {e+8}(e>=8);
// lanes 16-31 -> M=lane-16, K offset +8.  Two contiguous 16B loads per lane.
__device__ __forceinline__ v16bf load_a_frag(const unsigned short* A, int lda,
                                             int m0, int k0, int lane) {
  int r  = lane & 15;
  int kb = (lane >> 4) << 3;          // 0 or 8
  const unsigned short* p = A + (size_t)(m0 + r) * lda + k0;
  Frag16 f;
  f.q[0] = *(const uint4*)(p + kb);
  f.q[1] = *(const uint4*)(p + kb + 16);
  return f.v;
}

// B fragment: 32x16 bf16 (KxN), from a transposed [N][K] buffer.  lanes 0-15 -> N=lane,
// K={e}(e<8), {e+8}(e>=8); lanes 16-31 -> N=lane-16, K offset +8.  Two 16B loads per lane.
__device__ __forceinline__ v16bf load_b_frag(const unsigned short* Bt, int ldb,
                                             int n0, int k0, int lane) {
  int c  = lane & 15;
  int kb = (lane >> 4) << 3;          // 0 or 8
  const unsigned short* p = Bt + (size_t)(n0 + c) * ldb + k0 + kb;
  Frag16 f;
  f.q[0] = *(const uint4*)(p);
  f.q[1] = *(const uint4*)(p + 16);
  return f.v;
}

#define WMMA_BF16(a, b, c) \
  __builtin_amdgcn_wmma_f32_16x16x32_bf16(false, (a), false, (b), (short)0, (c), false, false)

// ---------------------------------------------------------------------------
// h[b,s,:] = emb[x[b,s],:] * sqrt(D) + pos[s,:]
__global__ void embed_kernel(const int* __restrict__ x, const float* __restrict__ emb,
                             const float* __restrict__ pos, float* __restrict__ h) {
  size_t i = (size_t)blockIdx.x * blockDim.x + threadIdx.x;   // float4 groups
  size_t n = (size_t)BB * SS * DD / 4;
  if (i >= n) return;
  size_t d4 = i % (DD / 4);
  size_t m  = i / (DD / 4);
  int s     = (int)(m % SS);
  int tok   = x[m];
  float4 e = ((const float4*)emb)[(size_t)tok * (DD / 4) + d4];
  float4 p = ((const float4*)pos)[(size_t)s   * (DD / 4) + d4];
  float4 r;
  r.x = e.x * 32.f + p.x;  r.y = e.y * 32.f + p.y;
  r.z = e.z * 32.f + p.z;  r.w = e.w * 32.f + p.w;
  ((float4*)h)[i] = r;
}

// Convert fp32 weight [K,N] -> bf16 transposed [N,K]
__global__ void trconv_kernel(const float* __restrict__ W, unsigned short* __restrict__ Wt,
                              int K, int N) {
  size_t i = (size_t)blockIdx.x * blockDim.x + threadIdx.x;
  size_t tot = (size_t)K * N;
  if (i >= tot) return;
  size_t k = i % K, n = i / K;
  Wt[i] = f2bf(W[k * (size_t)N + n]);
}

// LayerNorm row-wise over D, bf16 output
__global__ void ln_kernel(const float* __restrict__ in, const float* __restrict__ g,
                          const float* __restrict__ be, unsigned short* __restrict__ out) {
  int row = blockIdx.x, tid = threadIdx.x;
  const float* p = in + (size_t)row * DD;
  float s1 = 0.f, s2 = 0.f;
  for (int d = tid; d < DD; d += 256) { float v = p[d]; s1 += v; s2 += v * v; }
  __shared__ float r1[256], r2[256];
  r1[tid] = s1; r2[tid] = s2; __syncthreads();
  for (int off = 128; off > 0; off >>= 1) {
    if (tid < off) { r1[tid] += r1[tid + off]; r2[tid] += r2[tid + off]; }
    __syncthreads();
  }
  float mean = r1[0] * (1.f / DD);
  float var  = r2[0] * (1.f / DD) - mean * mean;
  float rstd = rsqrtf(var + 1e-5f);
  for (int d = tid; d < DD; d += 256)
    out[(size_t)row * DD + d] = f2bf((p[d] - mean) * rstd * g[d] + be[d]);
}

// ---------------------------------------------------------------------------
// LDS-tiled WMMA GEMM: C[M,N] = A[M,K](bf16) * Bt[N,K]^T(bf16) + bias.
// Block = 8 waves, block tile 128(M)x128(N), waves 4x2, wave tile 32x64.
// Per 32-wide K-step, A(128x32) and B(128x32) bf16 tiles are staged in LDS
// (rows padded to 40 halfs = 20 banks, 16 rows start in distinct 4-bank groups -> conflict-free frag reads).
// MODE: 0 = bf16 out, 1 = bf16 out written per-head-transposed (vT[b,h,d,s]),
//       2 = f32 out with residual add, 3 = relu bf16.
#define LDA_S 40
template <int MODE>
__global__ __launch_bounds__(256)
void wgemm_kernel(const unsigned short* __restrict__ A, const unsigned short* __restrict__ Bt,
                  const float* __restrict__ bias, const float* __restrict__ res,
                  float* __restrict__ outF, unsigned short* __restrict__ outBF,
                  int M, int N, int K) {
  __shared__ alignas(16) unsigned short As[128 * LDA_S];   // 10 KB
  __shared__ alignas(16) unsigned short Bs[128 * LDA_S];   // 10 KB

  int tid  = threadIdx.x;
  int lane = tid & 31;
  int wid  = tid >> 5;
  int nB = N >> 7;
  int m0 = (int)(blockIdx.x / nB) << 7;
  int n0 = (int)(blockIdx.x % nB) << 7;
  int wm = (wid >> 1) << 5;        // wave M offset within block: 0/32/64/96
  int wn = (wid & 1) << 6;         // wave N offset within block: 0/64

  // staging map: thread t -> tile row tr = t>>1, 32B segment th = (t&1)*16 halfs
  int tr = tid >> 1;
  int th = (tid & 1) << 4;

  v8f acc[8];
#pragma unroll
  for (int i = 0; i < 8; i++) acc[i] = zero8();

  const unsigned short* Arow = A  + (size_t)(m0 + tr) * K + th;
  const unsigned short* Brow = Bt + (size_t)(n0 + tr) * K + th;
  unsigned short* AsW = As + tr * LDA_S + th;
  unsigned short* BsW = Bs + tr * LDA_S + th;

  for (int k0 = 0; k0 < K; k0 += 32) {
    uint4 a0 = *(const uint4*)(Arow + k0);
    uint4 a1 = *(const uint4*)(Arow + k0 + 8);
    uint4 b0 = *(const uint4*)(Brow + k0);
    uint4 b1 = *(const uint4*)(Brow + k0 + 8);
    *(uint4*)(AsW)     = a0;
    *(uint4*)(AsW + 8) = a1;
    *(uint4*)(BsW)     = b0;
    *(uint4*)(BsW + 8) = b1;
    __syncthreads();

    v16bf af0 = load_a_frag(As, LDA_S, wm,      0, lane);
    v16bf af1 = load_a_frag(As, LDA_S, wm + 16, 0, lane);
#pragma unroll
    for (int j = 0; j < 4; j++) {
      v16bf bf = load_b_frag(Bs, LDA_S, wn + j * 16, 0, lane);
      acc[j]     = WMMA_BF16(af0, bf, acc[j]);
      acc[4 + j] = WMMA_BF16(af1, bf, acc[4 + j]);
    }
    __syncthreads();               // protect next iteration's staging stores
  }

  int cl = lane & 15;
#pragma unroll
  for (int h2 = 0; h2 < 2; h2++) {
    int rbase = m0 + wm + h2 * 16 + ((lane >> 4) << 3);
#pragma unroll
    for (int j = 0; j < 4; j++) {
      int col = n0 + wn + j * 16 + cl;
      float bcol = bias[col];
#pragma unroll
      for (int i = 0; i < 8; i++) {
        int row = rbase + i;
        float v = acc[h2 * 4 + j][i] + bcol;
        if (MODE == 0) {
          outBF[(size_t)row * N + col] = f2bf(v);
        } else if (MODE == 1) {        // vT[b,h,d,s]
          int b_ = row >> 10, s_ = row & 1023;
          int h_ = col >> 6,  d_ = col & 63;
          outBF[(((size_t)(b_ * HH + h_) * DHD + d_) << 10) | s_] = f2bf(v);
        } else if (MODE == 2) {
          outF[(size_t)row * N + col] = res[(size_t)row * N + col] + v;
        } else {                       // 3: relu
          outBF[(size_t)row * N + col] = f2bf(fmaxf(v, 0.f));
        }
      }
    }
  }
}

// ---------------------------------------------------------------------------
// Fused attention: one wave per (b, h, 16-query tile).
// scores (16x1024, f32) live in 64KB LDS; softmax in place; bf16 attn aliased into the
// same LDS (register-staged per row to avoid the alias hazard); ctx = attn @ vT via WMMA.
__global__ __launch_bounds__(32)
void attn_kernel(const unsigned short* __restrict__ q, const unsigned short* __restrict__ k,
                 const unsigned short* __restrict__ vT, const int* __restrict__ vlen,
                 unsigned short* __restrict__ ctx) {
  __shared__ alignas(16) float sc[16 * 1024];      // 64 KB
  unsigned short* ab = (unsigned short*)sc;        // bf16 attn alias (first 32 KB)

  int lane = threadIdx.x;
  int qt = blockIdx.x & 63;                        // S/16 = 64 tiles
  int bh = blockIdx.x >> 6;
  int b  = bh >> 4, h = bh & 15;
  int m0 = qt << 4;
  int vl = vlen[b];

  const unsigned short* qb = q + ((size_t)b * SS) * DD + (h << 6);
  const unsigned short* kb = k + ((size_t)b * SS) * DD + (h << 6);

  v16bf a0 = load_a_frag(qb, DD, m0,  0, lane);
  v16bf a1 = load_a_frag(qb, DD, m0, 32, lane);

  int rb = (lane >> 4) << 3;
  int cl = lane & 15;
  for (int n0 = 0; n0 < SS; n0 += 16) {
    v8f c = zero8();
    c = WMMA_BF16(a0, load_b_frag(kb, DD, n0,  0, lane), c);
    c = WMMA_BF16(a1, load_b_frag(kb, DD, n0, 32, lane), c);
    int col = n0 + cl;
    bool msk = (col >= vl);
#pragma unroll
    for (int i = 0; i < 8; i++)
      sc[(rb + i) * 1024 + col] = msk ? -1e9f : c[i] * 0.125f;
  }
  __syncthreads();

  for (int r = 0; r < 16; r++) {
    float* row = sc + r * 1024;
    float mx = -3.0e38f;
    for (int c2 = lane; c2 < 1024; c2 += 32) mx = fmaxf(mx, row[c2]);
    for (int off = 16; off; off >>= 1) mx = fmaxf(mx, __shfl_xor(mx, off, 32));
    float sum = 0.f;
    for (int c2 = lane; c2 < 1024; c2 += 32) {
      float e = __expf(row[c2] - mx);
      row[c2] = e; sum += e;
    }
    for (int off = 16; off; off >>= 1) sum += __shfl_xor(sum, off, 32);
    float inv = 1.f / sum;
    unsigned short st[32];
#pragma unroll
    for (int ii = 0; ii < 32; ii++) st[ii] = f2bf(row[lane + ii * 32] * inv);
    __syncthreads();                 // all float reads of this row done before alias write
#pragma unroll
    for (int ii = 0; ii < 32; ii++) ab[r * 1024 + lane + ii * 32] = st[ii];
    __syncthreads();
  }

  // ctx[16, 64] = attn[16, 1024] @ v[1024, 64]   (vT is [d, s] row-major per head)
  const unsigned short* vb = vT + ((size_t)bh * DHD) * SS;
  v8f cc0 = zero8(), cc1 = zero8(), cc2 = zero8(), cc3 = zero8();
  for (int k0 = 0; k0 < SS; k0 += 32) {
    v16bf a = load_a_frag(ab, 1024, 0, k0, lane);
    cc0 = WMMA_BF16(a, load_b_frag(vb, SS,  0, k0, lane), cc0);
    cc1 = WMMA_BF16(a, load_b_frag(vb, SS, 16, k0, lane), cc1);
    cc2 = WMMA_BF16(a, load_b_frag(vb, SS, 32, k0, lane), cc2);
    cc3 = WMMA_BF16(a, load_b_frag(vb, SS, 48, k0, lane), cc3);
  }
  v8f ccs[4] = {cc0, cc1, cc2, cc3};
#pragma unroll
  for (int j = 0; j < 4; j++)
#pragma unroll
    for (int i = 0; i < 8; i++) {
      size_t row = (size_t)b * SS + m0 + rb + i;
      int col = (h << 6) + j * 16 + cl;
      ctx[row * DD + col] = f2bf(ccs[j][i]);
    }
}

// ---------------------------------------------------------------------------
// Workspace layout (bytes, all 256-aligned by construction)
static constexpr size_t OFF_H   = 0;                         // f32  [8192,1024]  32MB
static constexpr size_t OFF_MID = OFF_H   + 33554432;        // f32  [8192,1024]  32MB
static constexpr size_t OFF_X1  = OFF_MID + 33554432;        // bf16 [8192,1024]  16MB
static constexpr size_t OFF_X2  = OFF_X1  + 16777216;        // bf16 [8192,1024]  16MB
static constexpr size_t OFF_Q   = OFF_X2  + 16777216;        // bf16 [8192,1024]  16MB
static constexpr size_t OFF_K   = OFF_Q   + 16777216;        // bf16 [8192,1024]  16MB
static constexpr size_t OFF_VT  = OFF_K   + 16777216;        // bf16 [B*H,64,1024] 16MB
static constexpr size_t OFF_CTX = OFF_VT  + 16777216;        // bf16 [8192,1024]  16MB
static constexpr size_t OFF_T   = OFF_CTX + 16777216;        // bf16 [8192,4096]  64MB
static constexpr size_t OFF_WQT = OFF_T   + 67108864;        // bf16 [1024,1024]   2MB
static constexpr size_t OFF_WKT = OFF_WQT + 2097152;
static constexpr size_t OFF_WVT = OFF_WKT + 2097152;
static constexpr size_t OFF_WOT = OFF_WVT + 2097152;
static constexpr size_t OFF_W1T = OFF_WOT + 2097152;         // bf16 [4096,1024]   8MB
static constexpr size_t OFF_W2T = OFF_W1T + 8388608;         // bf16 [1024,4096]   8MB

extern "C" void kernel_launch(void* const* d_in, const int* in_sizes, int n_in,
                              void* d_out, int out_size, void* d_ws, size_t ws_size,
                              hipStream_t stream) {
  (void)in_sizes; (void)n_in; (void)out_size; (void)ws_size;

  const int*   x    = (const int*)d_in[0];
  const int*   vlen = (const int*)d_in[1];
  const float* emb  = (const float*)d_in[2];
  const float* pos  = (const float*)d_in[3];
  const float* Wq   = (const float*)d_in[4]  + (size_t)LSEL * DD * DD;
  const float* bq   = (const float*)d_in[5]  + (size_t)LSEL * DD;
  const float* Wk   = (const float*)d_in[6]  + (size_t)LSEL * DD * DD;
  const float* bk   = (const float*)d_in[7]  + (size_t)LSEL * DD;
  const float* Wv   = (const float*)d_in[8]  + (size_t)LSEL * DD * DD;
  const float* bv   = (const float*)d_in[9]  + (size_t)LSEL * DD;
  const float* Wo   = (const float*)d_in[10] + (size_t)LSEL * DD * DD;
  const float* bo   = (const float*)d_in[11] + (size_t)LSEL * DD;
  const float* W1   = (const float*)d_in[12] + (size_t)LSEL * DD * FFD;
  const float* b1   = (const float*)d_in[13] + (size_t)LSEL * FFD;
  const float* W2   = (const float*)d_in[14] + (size_t)LSEL * FFD * DD;
  const float* b2   = (const float*)d_in[15] + (size_t)LSEL * DD;
  const float* g1   = (const float*)d_in[16] + (size_t)LSEL * DD;
  const float* be1  = (const float*)d_in[17] + (size_t)LSEL * DD;
  const float* g2   = (const float*)d_in[18] + (size_t)LSEL * DD;
  const float* be2  = (const float*)d_in[19] + (size_t)LSEL * DD;

  char* ws = (char*)d_ws;
  float*          h   = (float*)(ws + OFF_H);
  float*          mid = (float*)(ws + OFF_MID);
  unsigned short* x1  = (unsigned short*)(ws + OFF_X1);
  unsigned short* x2  = (unsigned short*)(ws + OFF_X2);
  unsigned short* qb  = (unsigned short*)(ws + OFF_Q);
  unsigned short* kb  = (unsigned short*)(ws + OFF_K);
  unsigned short* vT  = (unsigned short*)(ws + OFF_VT);
  unsigned short* ctx = (unsigned short*)(ws + OFF_CTX);
  unsigned short* t   = (unsigned short*)(ws + OFF_T);
  unsigned short* WqT = (unsigned short*)(ws + OFF_WQT);
  unsigned short* WkT = (unsigned short*)(ws + OFF_WKT);
  unsigned short* WvT = (unsigned short*)(ws + OFF_WVT);
  unsigned short* WoT = (unsigned short*)(ws + OFF_WOT);
  unsigned short* W1T = (unsigned short*)(ws + OFF_W1T);
  unsigned short* W2T = (unsigned short*)(ws + OFF_W2T);

  float* out = (float*)d_out;
  const int M = BB * SS;   // 8192

  // 0) embedding + positional encoding
  embed_kernel<<<(M * DD / 4 + 255) / 256, 256, 0, stream>>>(x, emb, pos, h);

  // 1) weight convert + transpose (bf16, [N,K])
  trconv_kernel<<<4096, 256, 0, stream>>>(Wq, WqT, DD, DD);
  trconv_kernel<<<4096, 256, 0, stream>>>(Wk, WkT, DD, DD);
  trconv_kernel<<<4096, 256, 0, stream>>>(Wv, WvT, DD, DD);
  trconv_kernel<<<4096, 256, 0, stream>>>(Wo, WoT, DD, DD);
  trconv_kernel<<<16384, 256, 0, stream>>>(W1, W1T, DD, FFD);   // [D,FF] -> [FF,D]
  trconv_kernel<<<16384, 256, 0, stream>>>(W2, W2T, FFD, DD);   // [FF,D] -> [D,FF]

  // 2) LN1
  ln_kernel<<<M, 256, 0, stream>>>(h, g1, be1, x1);

  // 3) Q, K, V projections (V written per-head transposed)
  int gqkv = (M / 128) * (DD / 128);     // 512 blocks
  wgemm_kernel<0><<<gqkv, 256, 0, stream>>>(x1, WqT, bq, nullptr, nullptr, qb, M, DD, DD);
  wgemm_kernel<0><<<gqkv, 256, 0, stream>>>(x1, WkT, bk, nullptr, nullptr, kb, M, DD, DD);
  wgemm_kernel<1><<<gqkv, 256, 0, stream>>>(x1, WvT, bv, nullptr, nullptr, vT, M, DD, DD);

  // 4) attention (8*16 heads * 64 q-tiles = 8192 waves)
  attn_kernel<<<BB * HH * (SS / 16), 32, 0, stream>>>(qb, kb, vT, vlen, ctx);

  // 5) mid = h + ctx @ Wo + bo
  wgemm_kernel<2><<<gqkv, 256, 0, stream>>>(ctx, WoT, bo, h, mid, nullptr, M, DD, DD);

  // 6) LN2
  ln_kernel<<<M, 256, 0, stream>>>(mid, g2, be2, x2);

  // 7) t = relu(x2 @ W1 + b1)
  int gff1 = (M / 128) * (FFD / 128);    // 2048 blocks
  wgemm_kernel<3><<<gff1, 256, 0, stream>>>(x2, W1T, b1, nullptr, nullptr, t, M, FFD, DD);

  // 8) out = mid + t @ W2 + b2
  wgemm_kernel<2><<<gqkv, 256, 0, stream>>>(t, W2T, b2, mid, out, nullptr, M, DD, FFD);
}
